// MultiheadAttentionPlus_18013092839747
// MI455X (gfx1250) — matrix operand order repaired and checked by reading, not compiled, not run on hardware
//
#include <hip/hip_runtime.h>
#include <hip/hip_bf16.h>

// ---- problem constants (reference: L=2048, B=2, E=1024, H=16, D=64) ----
#define LQ   2048
#define NB   2
#define EE   1024
#define HH   16
#define DD   64
#define NN   (LQ * NB)          // 4096 rows of x
#define SCAL 0.125f             // D^-0.5

typedef __bf16 bf16;
typedef bf16  v16bf __attribute__((ext_vector_type(16)));
typedef float v8f   __attribute__((ext_vector_type(8)));

union AFrag {
    v16bf v;
    bf16  e[16];
    uint4 q[2];
};

__device__ inline v8f wmma_bf16(const AFrag& a, const AFrag& b, v8f c) {
    return __builtin_amdgcn_wmma_f32_16x16x32_bf16(false, a.v, false, b.v,
                                                   (short)0, c, false, false);
}

// A-matrix fragment (16x32 bf16), contiguous-K layout (same order as load_b):
// lane m=lane&15 holds row M=m; element e: K = kb + ((lane>>4)<<4) + e  (16 contiguous)
__device__ inline AFrag load_a(const bf16* base, int rowStride, int row16, int kb, int lane) {
    AFrag f;
    const int m   = lane & 15;
    const int hi8 = (lane >> 4) << 4;
    const bf16* p = base + (size_t)(row16 + m) * rowStride + kb + hi8;
    f.q[0] = *(const uint4*)(p);        // elements 0..7  -> K = kb+hi16+0..7
    f.q[1] = *(const uint4*)(p + 8);    // elements 8..15 -> K = kb+hi16+8..15
    return f;
}

// B-matrix fragment (32x16 bf16), contiguous-K layout (same order as load_a):
// lane holds column n=lane&15; element e: K = kb + ((lane>>4)<<4) + e  (16 contiguous)
// `base` is indexed [n][K] with stride rowStride (i.e. B = base^T).
__device__ inline AFrag load_b(const bf16* base, int rowStride, int col16, int kb, int lane) {
    AFrag f;
    const int n   = lane & 15;
    const int klo = (lane >> 4) << 4;
    const bf16* p = base + (size_t)(col16 + n) * rowStride + kb + klo;
    f.q[0] = *(const uint4*)(p);
    f.q[1] = *(const uint4*)(p + 8);
    return f;
}

// B fragment from a transposed [n][K] LDS tile with K-stride 32 (contraction contiguous).
__device__ inline AFrag load_b_lds_t(const bf16* baseT, int col16, int lane) {
    AFrag f;
    const bf16* p = baseT + (size_t)(col16 + (lane & 15)) * 32 + ((lane >> 4) << 4);
    f.q[0] = *(const uint4*)(p);
    f.q[1] = *(const uint4*)(p + 8);
    return f;
}

// ---------------- kernel 0: fp32 -> bf16 (8 elements / thread) ----------------
__global__ __launch_bounds__(256) void f32_to_bf16_kernel(const float* __restrict__ in,
                                                          bf16* __restrict__ out, int n8) {
    int i = blockIdx.x * blockDim.x + threadIdx.x;
    if (i >= n8) return;
    const float* p = in + (size_t)i * 8;
    union { bf16 e[8]; uint4 q; } u;
#pragma unroll
    for (int j = 0; j < 8; ++j) u.e[j] = (bf16)p[j];
    ((uint4*)out)[i] = u.q;
}

// ---------------- kernel 1: fused Q/K/V projection --------------------------
// One wave computes a 32x64 output tile of  x(4096x1024) @ W^T(1024x1024):
// 2 A-frags x 4 B-frags -> 8 WMMAs per K-step.  All fragment loads for a
// K-step are issued before the WMMAs so VMEM overlaps the matrix pipe.
__global__ __launch_bounds__(128) void qkv_gemm(const bf16* __restrict__ xb,
                                                const bf16* __restrict__ Wqb,
                                                const bf16* __restrict__ Wkb,
                                                const bf16* __restrict__ Wvb,
                                                const float* __restrict__ bq,
                                                const float* __restrict__ bk,
                                                const float* __restrict__ bv,
                                                bf16* __restrict__ q_ws,
                                                bf16* __restrict__ k_ws,
                                                bf16* __restrict__ v_ws) {
    const int lane = threadIdx.x & 31;
    const int wid  = blockIdx.x * 4 + (threadIdx.x >> 5);
    const int proj = wid >> 11;          // 2048 = 128 mtiles * 16 ntiles
    const int rem  = wid & 2047;
    const int mt   = rem >> 4;           // 32-row tile index
    const int nt64 = rem & 15;

    const bf16*  W    = (proj == 0) ? Wqb : (proj == 1) ? Wkb : Wvb;
    const float* bias = (proj == 0) ? bq  : (proj == 1) ? bk  : bv;
    bf16*        outp = (proj == 0) ? q_ws : (proj == 1) ? k_ws : v_ws;
    const float  scale = (proj == 0) ? SCAL : 1.0f;

    v8f acc[2][4];
#pragma unroll
    for (int mr = 0; mr < 2; ++mr)
#pragma unroll
        for (int nt = 0; nt < 4; ++nt)
#pragma unroll
            for (int r = 0; r < 8; ++r) acc[mr][nt][r] = 0.0f;

#pragma unroll 2
    for (int kb = 0; kb < EE; kb += 32) {
        AFrag a0 = load_a(xb, EE, mt * 32,      kb, lane);
        AFrag a1 = load_a(xb, EE, mt * 32 + 16, kb, lane);
        AFrag b[4];
#pragma unroll
        for (int nt = 0; nt < 4; ++nt)
            b[nt] = load_b(W, EE, nt64 * 64 + nt * 16, kb, lane);
#pragma unroll
        for (int nt = 0; nt < 4; ++nt) {
            acc[0][nt] = wmma_bf16(a0, b[nt], acc[0][nt]);
            acc[1][nt] = wmma_bf16(a1, b[nt], acc[1][nt]);
        }
    }

    const int hi  = lane >> 4;
    const int nlo = lane & 15;
#pragma unroll
    for (int nt = 0; nt < 4; ++nt) {
        const int e = nt64 * 64 + nt * 16 + nlo;
        const int h = e >> 6, d = e & 63;
        const float be = bias[e];
#pragma unroll
        for (int mr = 0; mr < 2; ++mr)
#pragma unroll
            for (int r = 0; r < 8; ++r) {
                const int n  = mt * 32 + mr * 16 + r + hi * 8;   // row in (L*B)
                const int l  = n >> 1;                           // B == 2
                const int bb = n & 1;
                const float val = (acc[mr][nt][r] + be) * scale;
                outp[((size_t)(bb * HH + h) * LQ + l) * DD + d] = (bf16)val;
            }
    }
}

// ---------------- kernel 2: flash attention ----------------------------------
// 1 block = 1 head x 64 queries (4 waves x 16 queries). Key blocks of 32 staged
// in LDS (V stored transposed so P.V fragments are b128 LDS reads); online
// softmax; P transposed via per-wave LDS scratch into A-fragment layout.
__global__ __launch_bounds__(128) void attn_kernel(const bf16* __restrict__ q_ws,
                                                   const bf16* __restrict__ k_ws,
                                                   const bf16* __restrict__ v_ws,
                                                   bf16* __restrict__ ctx) {
    __shared__ bf16 Klds[32 * DD];       // [key][d]   4 KB
    __shared__ bf16 Vlds[DD * 32];       // [d][key]   4 KB (transposed)
    __shared__ bf16 Plds[4 * 16 * 32];   // 4 KB (1 KB per wave)

    const int tid  = threadIdx.x;
    const int lane = tid & 31;
    const int w    = tid >> 5;
    const int head = blockIdx.x >> 5;
    const int qblk = blockIdx.x & 31;
    const int qbase = qblk * 64 + w * 16;

    const bf16* Qh = q_ws + (size_t)head * LQ * DD;
    const bf16* Kh = k_ws + (size_t)head * LQ * DD;
    const bf16* Vh = v_ws + (size_t)head * LQ * DD;

    AFrag qf0 = load_a(Qh, DD, qbase, 0, lane);
    AFrag qf1 = load_a(Qh, DD, qbase, 32, lane);

    float m[8], lsum[8];
    v8f o[4];
#pragma unroll
    for (int r = 0; r < 8; ++r) { m[r] = -1e30f; lsum[r] = 0.0f; }
#pragma unroll
    for (int nt = 0; nt < 4; ++nt)
#pragma unroll
        for (int r = 0; r < 8; ++r) o[nt][r] = 0.0f;

    const int hi  = lane >> 4;
    const int nlo = lane & 15;
    bf16* Pw = Plds + w * 512;

    v8f zero;
#pragma unroll
    for (int r = 0; r < 8; ++r) zero[r] = 0.0f;

    // staging coords: K straight copy, V transposed scatter
    const int vk = tid >> 2;             // key row 0..31
    const int vd = (tid & 3) * 16;       // d group

    for (int kb = 0; kb < LQ; kb += 32) {
        __syncthreads();
        {   // K block: straight uint4 copy (32 keys x 64 d bf16 = 256 uint4)
            const uint4* Ks = (const uint4*)(Kh + (size_t)kb * DD);
            uint4* Kd = (uint4*)Klds;
            Kd[tid]       = Ks[tid];
            Kd[tid + 128] = Ks[tid + 128];
            // V block: transpose into [d][key]
            AFrag u;
            const bf16* vs = Vh + (size_t)(kb + vk) * DD + vd;
            u.q[0] = *(const uint4*)(vs);
            u.q[1] = *(const uint4*)(vs + 8);
#pragma unroll
            for (int j = 0; j < 16; ++j)
                Vlds[(vd + j) * 32 + vk] = u.e[j];
        }
        __syncthreads();

        if (kb + 32 < LQ) {   // hint next block into cache (global_prefetch_b8)
            __builtin_prefetch(Kh + (size_t)(kb + 32) * DD + tid * 16, 0, 1);
            __builtin_prefetch(Vh + (size_t)(kb + 32) * DD + tid * 16, 0, 1);
        }

        // scores: S(16q x 32k) = Q(16x64) . K^T -> two 16x16 tiles, 2 WMMAs each
        AFrag b00 = load_b(Klds, DD, 0,  0,  lane);
        AFrag b01 = load_b(Klds, DD, 0,  32, lane);
        AFrag b10 = load_b(Klds, DD, 16, 0,  lane);
        AFrag b11 = load_b(Klds, DD, 16, 32, lane);
        v8f s0 = wmma_bf16(qf0, b00, zero);
        v8f s1 = wmma_bf16(qf0, b10, zero);
        s0     = wmma_bf16(qf1, b01, s0);
        s1     = wmma_bf16(qf1, b11, s1);

        // online softmax (rows r+hi*8 live in this lane's 16-lane half)
#pragma unroll
        for (int r = 0; r < 8; ++r) {
            float vmax = fmaxf(s0[r], s1[r]);
#pragma unroll
            for (int off = 1; off < 16; off <<= 1)
                vmax = fmaxf(vmax, __shfl_xor(vmax, off, 32));
            const float mn = fmaxf(m[r], vmax);
            const float sc = __expf(m[r] - mn);
            const float p0 = __expf(s0[r] - mn);
            const float p1 = __expf(s1[r] - mn);
            float rs = p0 + p1;
#pragma unroll
            for (int off = 1; off < 16; off <<= 1)
                rs += __shfl_xor(rs, off, 32);
            lsum[r] = lsum[r] * sc + rs;
            m[r] = mn;
#pragma unroll
            for (int nt = 0; nt < 4; ++nt) o[nt][r] *= sc;
            const int row = r + hi * 8;
            Pw[row * 32 + nlo]      = (bf16)p0;
            Pw[row * 32 + 16 + nlo] = (bf16)p1;
        }

        // P(16x32) . V(32x64): P in A layout, V fragments are b128 LDS reads
        AFrag pf = load_a(Pw, 32, 0, 0, lane);
        AFrag vf[4];
#pragma unroll
        for (int nt = 0; nt < 4; ++nt)
            vf[nt] = load_b_lds_t(Vlds, nt * 16, lane);
#pragma unroll
        for (int nt = 0; nt < 4; ++nt)
            o[nt] = wmma_bf16(pf, vf[nt], o[nt]);
    }

    // epilogue: ctx laid out (L*B, E) row-major, e = h*64 + d
    const int bb = head >> 4;     // head = b*H + h
    const int h  = head & 15;
#pragma unroll
    for (int nt = 0; nt < 4; ++nt) {
        const int e = h * 64 + nt * 16 + nlo;
#pragma unroll
        for (int r = 0; r < 8; ++r) {
            const int l = qbase + r + hi * 8;
            ctx[((size_t)(l * NB + bb)) * EE + e] = (bf16)(o[nt][r] / lsum[r]);
        }
    }
}

// ---------------- kernel 3: output projection (32x64 tile / wave) ------------
__global__ __launch_bounds__(128) void out_gemm(const bf16* __restrict__ ctx,
                                                const bf16* __restrict__ Wob,
                                                const float* __restrict__ bo,
                                                float* __restrict__ out) {
    const int lane = threadIdx.x & 31;
    const int wid  = blockIdx.x * 4 + (threadIdx.x >> 5);
    const int mt   = wid >> 4;           // 128 tiles of 32 rows
    const int nt64 = wid & 15;

    v8f acc[2][4];
#pragma unroll
    for (int mr = 0; mr < 2; ++mr)
#pragma unroll
        for (int nt = 0; nt < 4; ++nt)
#pragma unroll
            for (int r = 0; r < 8; ++r) acc[mr][nt][r] = 0.0f;

#pragma unroll 2
    for (int kb = 0; kb < EE; kb += 32) {
        AFrag a0 = load_a(ctx, EE, mt * 32,      kb, lane);
        AFrag a1 = load_a(ctx, EE, mt * 32 + 16, kb, lane);
        AFrag b[4];
#pragma unroll
        for (int nt = 0; nt < 4; ++nt)
            b[nt] = load_b(Wob, EE, nt64 * 64 + nt * 16, kb, lane);
#pragma unroll
        for (int nt = 0; nt < 4; ++nt) {
            acc[0][nt] = wmma_bf16(a0, b[nt], acc[0][nt]);
            acc[1][nt] = wmma_bf16(a1, b[nt], acc[1][nt]);
        }
    }

    const int hi  = lane >> 4;
    const int nlo = lane & 15;
#pragma unroll
    for (int nt = 0; nt < 4; ++nt) {
        const int e = nt64 * 64 + nt * 16 + nlo;
        const float be = bo[e];
#pragma unroll
        for (int mr = 0; mr < 2; ++mr)
#pragma unroll
            for (int r = 0; r < 8; ++r) {
                const int n = mt * 32 + mr * 16 + r + hi * 8;
                out[(size_t)n * EE + e] = acc[mr][nt][r] + be;
            }
    }
}

// ---------------- host-side launcher -----------------------------------------
extern "C" void kernel_launch(void* const* d_in, const int* in_sizes, int n_in,
                              void* d_out, int out_size, void* d_ws, size_t ws_size,
                              hipStream_t stream) {
    const float* x  = (const float*)d_in[0];
    const float* Wq = (const float*)d_in[1];
    const float* bq = (const float*)d_in[2];
    const float* Wk = (const float*)d_in[3];
    const float* bk = (const float*)d_in[4];
    const float* Wv = (const float*)d_in[5];
    const float* bv = (const float*)d_in[6];
    const float* Wo = (const float*)d_in[7];
    const float* bo = (const float*)d_in[8];
    float* out = (float*)d_out;

    char* ws = (char*)d_ws;
    const size_t MB = 1u << 20;
    bf16* xb   = (bf16*)(ws + 0 * MB);    // 8 MB  (4096x1024)
    bf16* Wqb  = (bf16*)(ws + 8 * MB);    // 2 MB
    bf16* Wkb  = (bf16*)(ws + 10 * MB);   // 2 MB
    bf16* Wvb  = (bf16*)(ws + 12 * MB);   // 2 MB
    bf16* Wob  = (bf16*)(ws + 14 * MB);   // 2 MB
    bf16* q_ws = (bf16*)(ws + 16 * MB);   // 8 MB  (32, 2048, 64)
    bf16* k_ws = (bf16*)(ws + 24 * MB);   // 8 MB
    bf16* v_ws = (bf16*)(ws + 32 * MB);   // 8 MB
    bf16* ctxw = (bf16*)(ws + 40 * MB);   // 8 MB  (4096x1024)  -> total 48 MB

    const int n8x = NN * EE / 8;          // 524288
    const int n8w = EE * EE / 8;          // 131072
    f32_to_bf16_kernel<<<n8x / 256, 256, 0, stream>>>(x,  xb,  n8x);
    f32_to_bf16_kernel<<<n8w / 256, 256, 0, stream>>>(Wq, Wqb, n8w);
    f32_to_bf16_kernel<<<n8w / 256, 256, 0, stream>>>(Wk, Wkb, n8w);
    f32_to_bf16_kernel<<<n8w / 256, 256, 0, stream>>>(Wv, Wvb, n8w);
    f32_to_bf16_kernel<<<n8w / 256, 256, 0, stream>>>(Wo, Wob, n8w);

    // 3 projections * 128 mtiles(32 rows) * 16 ntiles = 6144 waves, 4/block
    qkv_gemm<<<1536, 128, 0, stream>>>(xb, Wqb, Wkb, Wvb, bq, bk, bv,
                                       q_ws, k_ws, v_ws);

    // 32 heads * 32 query blocks
    attn_kernel<<<1024, 128, 0, stream>>>(q_ws, k_ws, v_ws, ctxw);

    // 128 mtiles(32 rows) * 16 ntiles = 2048 waves, 4/block
    out_gemm<<<512, 128, 0, stream>>>(ctxw, Wob, bo, out);
}
